// Transformer_69114613727480
// MI455X (gfx1250) — matrix operand order repaired and checked by reading, not compiled, not run on hardware
//
#include <hip/hip_runtime.h>

// ---------------------------------------------------------------------------
// GPT-style transformer forward (B=8,T=2048,C=512,H=8,NL=2) for gfx1250.
// bf16 WMMA (v_wmma_f32_16x16x32_bf16), fp32 accumulation, TDM (tensor DMA)
// double-buffered LDS staging in the GEMMs, flash-attention w/ online softmax.
// ---------------------------------------------------------------------------

typedef unsigned short bf16_t;
typedef __bf16 v16bf __attribute__((ext_vector_type(16)));
typedef float  v8f   __attribute__((ext_vector_type(8)));
typedef unsigned int u32x4 __attribute__((ext_vector_type(4)));
typedef int i32x8 __attribute__((ext_vector_type(8)));
typedef int i32x4 __attribute__((ext_vector_type(4)));

#define B_  8
#define T_  2048
#define C_  512
#define H_  8
#define HD_ 64
#define M_  (B_ * T_)
#define TQ_ (T_ / 16)

// fp32 -> bf16 round-to-nearest-even
__device__ __forceinline__ bf16_t f2bf(float f) {
  union { float f; unsigned u; } x; x.f = f;
  unsigned r = x.u + 0x7FFFu + ((x.u >> 16) & 1u);
  return (bf16_t)(r >> 16);
}

union FragU { v16bf v; uint4 u[2]; };

// Load a 16-element bf16 operand fragment as two 16B (global or LDS) loads.
__device__ __forceinline__ v16bf ldfrag(const bf16_t* p0, const bf16_t* p1) {
  FragU f;
  f.u[0] = *reinterpret_cast<const uint4*>(p0);
  f.u[1] = *reinterpret_cast<const uint4*>(p1);
  return f.v;
}

__device__ __forceinline__ v8f zero8() {
  v8f z;
#pragma unroll
  for (int i = 0; i < 8; ++i) z[i] = 0.f;
  return z;
}

__device__ __forceinline__ v8f wmma_bf16(v16bf a, v16bf b, v8f c) {
  return __builtin_amdgcn_wmma_f32_16x16x32_bf16(
      /*neg_a=*/false, a, /*neg_b=*/false, b,
      /*c_mod=*/(short)0, c, /*reuse_a=*/false, /*reuse_b=*/false);
}

// ---------------------------------------------------------------------------
// Tensor Data Mover: 2-D bf16 tile load, global -> LDS (D# per ISA ch.8).
// tile_h rows of tile_w elements, source row stride = stride_elems.
// Groups 2/3 zero (<=2D tensor); workgroup_mask=0 (not in a cluster).
// 6-arg builtin form (amdgpu-toolchain clang-23): extra int32x8, zero-filled.
// ---------------------------------------------------------------------------
__device__ __forceinline__ void tdm_load_2d(const bf16_t* gptr, bf16_t* lptr,
                                            unsigned tile_w, unsigned tile_h,
                                            unsigned stride_elems,
                                            unsigned tensor_w,
                                            unsigned tensor_h) {
  const unsigned long long ga = (unsigned long long)gptr;
  const unsigned lds = (unsigned)(unsigned long long)lptr;  // LDS byte offset
  u32x4 g0;
  g0[0] = 1u;                                   // count=1, user descriptor
  g0[1] = lds;                                  // lds_addr
  g0[2] = (unsigned)ga;                         // global_addr[31:0]
  g0[3] = (unsigned)(ga >> 32) | (2u << 30);    // global_addr[56:32] | type=2
  i32x8 g1;
  g1[0] = (int)(1u << 16);                      // data_size=1 (2 bytes)
  g1[1] = (int)((tensor_w & 0xFFFFu) << 16);    // tensor_dim0[15:0]
  g1[2] = (int)((tensor_w >> 16) | ((tensor_h & 0xFFFFu) << 16));
  g1[3] = (int)((tensor_h >> 16) | (tile_w << 16));  // tile_dim0
  g1[4] = (int)(tile_h & 0xFFFFu);              // tile_dim1 (tile_dim2=0)
  g1[5] = (int)stride_elems;                    // tensor_dim0_stride[31:0]
  g1[6] = 0;
  g1[7] = 0;
  i32x4 g2; g2[0] = g2[1] = g2[2] = g2[3] = 0;
  i32x4 g3; g3[0] = g3[1] = g3[2] = g3[3] = 0;
  i32x8 g4;
#pragma unroll
  for (int i = 0; i < 8; ++i) g4[i] = 0;
  __builtin_amdgcn_tensor_load_to_lds(g0, g1, g2, g3, g4, 0);
}

// ---------------------------------------------------------------------------
// Weight pack: fp32 [K,N] row-major -> bf16 tiles in WMMA B-operand order.
// Tile (kt,nt): lane l: n = nt*16+(l&15), k = kt*32+(l>>4)*16+j (j=0..15).
// ---------------------------------------------------------------------------
__global__ __launch_bounds__(256) void pack_w_k(const float* __restrict__ W,
                                                bf16_t* __restrict__ outp,
                                                int N, int total) {
  int idx = blockIdx.x * 256 + threadIdx.x;
  if (idx >= total) return;
  const int j    = idx & 15;
  const int lane = (idx >> 4) & 31;
  const int tile = idx >> 9;
  const int ntiles = N >> 4;
  const int kt = tile / ntiles;
  const int nt = tile - kt * ntiles;
  const int k = kt * 32 + (lane >> 4) * 16 + j;
  const int n = nt * 16 + (lane & 15);
  outp[idx] = f2bf(W[(size_t)k * N + n]);
}

// ---------------------------------------------------------------------------
// GEMM with TDM double-buffered LDS staging.
// Per K-step the TDM streams: A tile 128x32 (strided rows) and the packed-B
// panel (8 contiguous 16x32 tiles = 8KB). Waves consume via ds_load_b128.
// EPI 0: bf16 = acc+bias | 1: bf16 = gelu(acc+bias) | 2: f32 = resid+acc+bias
// ---------------------------------------------------------------------------
template <int EPI>
__global__ __launch_bounds__(128) void gemm_wmma_k(
    const bf16_t* __restrict__ A, const bf16_t* __restrict__ Bp,
    const float* __restrict__ bias, const float* __restrict__ resid,
    bf16_t* __restrict__ outB, float* __restrict__ outF,
    int Ndim, int Kdim) {
  __shared__ bf16_t lA[2][128 * 32];  // 2 x 8KB
  __shared__ bf16_t lB[2][8 * 512];   // 2 x 8KB
  const int lane = threadIdx.x & 31;
  const int wv   = threadIdx.x >> 5;
  const int l15  = lane & 15;
  const int lh   = lane >> 4;
  const int mblk = blockIdx.y * 128;
  const int nblk16 = blockIdx.x * 8;          // first n-tile of block
  const int mwl = (wv >> 1) * 64;             // wave M offset inside block
  const int jb  = (wv & 1) * 4;               // wave n-tile offset inside block
  const int ntiles = Ndim >> 4;
  const int nIt = Kdim >> 5;

  v8f acc[4][4];
#pragma unroll
  for (int i = 0; i < 4; ++i)
#pragma unroll
    for (int j = 0; j < 4; ++j) acc[i][j] = zero8();

  if (wv == 0) {  // prologue: stream K-step 0
    tdm_load_2d(A + (size_t)mblk * Kdim, lA[0], 32u, 128u, (unsigned)Kdim,
                (unsigned)Kdim, (unsigned)M_);
    tdm_load_2d(Bp + (size_t)nblk16 * 512, lB[0], 4096u, 1u, 4096u,
                (unsigned)(ntiles * (Kdim >> 5) * 512), 1u);
  }

  for (int it = 0; it < nIt; ++it) {
    __syncthreads();  // all waves done reading buffer (it+1)&1 from it-1
    if (wv == 0) {
      if (it + 1 < nIt) {
        const int kb1 = (it + 1) * 32;
        tdm_load_2d(A + (size_t)mblk * Kdim + kb1, lA[(it + 1) & 1], 32u, 128u,
                    (unsigned)Kdim, (unsigned)Kdim, (unsigned)M_);
        tdm_load_2d(Bp + ((size_t)(kb1 >> 5) * ntiles + nblk16) * 512,
                    lB[(it + 1) & 1], 4096u, 1u, 4096u,
                    (unsigned)(ntiles * (Kdim >> 5) * 512), 1u);
        __builtin_amdgcn_s_wait_tensorcnt((short)2);  // step `it` retired
      } else {
        __builtin_amdgcn_s_wait_tensorcnt((short)0);
      }
    }
    __syncthreads();  // buffer it&1 ready for everyone

    const bf16_t* Abuf = lA[it & 1];
    const bf16_t* Bbuf = lB[it & 1];
    v16bf af[4], bfr[4];
#pragma unroll
    for (int i = 0; i < 4; ++i) {
      // A-operand: lane<16: row l15, K {0..15}; lane>=16: {16..31}
      const bf16_t* ap = Abuf + (mwl + i * 16 + l15) * 32 + lh * 16;
      af[i] = ldfrag(ap, ap + 8);
    }
#pragma unroll
    for (int j = 0; j < 4; ++j) {
      const bf16_t* bp = Bbuf + (jb + j) * 512 + lane * 16;
      bfr[j] = ldfrag(bp, bp + 8);
    }
#pragma unroll
    for (int i = 0; i < 4; ++i)
#pragma unroll
      for (int j = 0; j < 4; ++j)
        acc[i][j] = wmma_bf16(af[i], bfr[j], acc[i][j]);
  }

  const int mw = mblk + mwl;
  const int nw = (nblk16 + jb) * 16;
#pragma unroll
  for (int i = 0; i < 4; ++i) {
#pragma unroll
    for (int j = 0; j < 4; ++j) {
      const int col = nw + j * 16 + l15;
      const float bv = bias[col];
#pragma unroll
      for (int v = 0; v < 8; ++v) {
        const int row = mw + i * 16 + v + 8 * lh;
        float x = acc[i][j][v] + bv;
        const size_t o = (size_t)row * Ndim + col;
        if (EPI == 1) {
          x = 0.5f * x *
              (1.f + tanhf(0.7978845608f * (x + 0.044715f * x * x * x)));
        }
        if (EPI == 2) {
          outF[o] = resid[o] + x;
        } else {
          outB[o] = f2bf(x);
        }
      }
    }
  }
}

// ---------------------------------------------------------------------------
// Flash attention, causal. One wave per 16-query tile per (b,h).
// ---------------------------------------------------------------------------
__global__ __launch_bounds__(128) void attn_flash_k(
    const bf16_t* __restrict__ Q, const bf16_t* __restrict__ Kmat,
    const bf16_t* __restrict__ VT, bf16_t* __restrict__ Y) {
  __shared__ bf16_t pls[4][16 * 32];  // per-wave P staging (16 rows x 32 keys)
  const int lane = threadIdx.x & 31;
  const int wv   = threadIdx.x >> 5;
  const int l15  = lane & 15;
  const int lh   = lane >> 4;
  const int w   = blockIdx.x * 4 + wv;
  const int b   = w / (H_ * TQ_);
  const int rem = w % (H_ * TQ_);
  const int h   = rem / TQ_;
  const int qt  = rem % TQ_;

  v16bf qf[2];
  const bf16_t* qbase = Q + ((size_t)b * T_ + qt * 16 + l15) * C_ + h * HD_;
#pragma unroll
  for (int kk = 0; kk < 2; ++kk) {
    const bf16_t* qp = qbase + kk * 32 + lh * 16;
    qf[kk] = ldfrag(qp, qp + 8);
  }

  v8f o[4];
#pragma unroll
  for (int t = 0; t < 4; ++t) o[t] = zero8();
  float mrow[8], lrow[8];
#pragma unroll
  for (int v = 0; v < 8; ++v) { mrow[v] = -1e30f; lrow[v] = 0.f; }

  const int nK = qt + 1;
  const int nB = (nK + 1) >> 1;
  const bf16_t* kbh = Kmat + (size_t)b * T_ * C_ + h * HD_;
  const bf16_t* vbh = VT + (size_t)(b * H_ + h) * HD_ * T_;

  for (int j2 = 0; j2 < nB; ++j2) {
    const int c0 = 2 * j2, c1 = c0 + 1;
    const bool has1 = (c1 < nK);
    float s0[8], s1[8];
    {
      const bf16_t* kp = kbh + (size_t)(c0 * 16 + l15) * C_ + lh * 16;
      v16bf kf0 = ldfrag(kp, kp + 8);
      v16bf kf1 = ldfrag(kp + 32, kp + 40);
      v8f s = zero8();
      s = wmma_bf16(qf[0], kf0, s);
      s = wmma_bf16(qf[1], kf1, s);
#pragma unroll
      for (int v = 0; v < 8; ++v) {
        float sv = s[v] * 0.125f;
        if (c0 == qt && l15 > (v + 8 * lh)) sv = -1e30f;
        s0[v] = sv;
      }
    }
    if (has1) {
      const bf16_t* kp = kbh + (size_t)(c1 * 16 + l15) * C_ + lh * 16;
      v16bf kf0 = ldfrag(kp, kp + 8);
      v16bf kf1 = ldfrag(kp + 32, kp + 40);
      v8f s = zero8();
      s = wmma_bf16(qf[0], kf0, s);
      s = wmma_bf16(qf[1], kf1, s);
#pragma unroll
      for (int v = 0; v < 8; ++v) {
        float sv = s[v] * 0.125f;
        if (c1 == qt && l15 > (v + 8 * lh)) sv = -1e30f;
        s1[v] = sv;
      }
    } else {
#pragma unroll
      for (int v = 0; v < 8; ++v) s1[v] = -1e30f;
    }

#pragma unroll
    for (int v = 0; v < 8; ++v) {
      float cm = fmaxf(s0[v], s1[v]);
#pragma unroll
      for (int mm = 8; mm >= 1; mm >>= 1) cm = fmaxf(cm, __shfl_xor(cm, mm));
      const float newm = fmaxf(mrow[v], cm);
      const float alpha = __expf(mrow[v] - newm);
      mrow[v] = newm;
      const float p0 = __expf(s0[v] - newm);
      const float p1 = has1 ? __expf(s1[v] - newm) : 0.f;
      float rs = p0 + p1;
#pragma unroll
      for (int mm = 8; mm >= 1; mm >>= 1) rs += __shfl_xor(rs, mm);
      lrow[v] = lrow[v] * alpha + rs;
#pragma unroll
      for (int t = 0; t < 4; ++t) o[t][v] *= alpha;
      const int r = v + 8 * lh;
      pls[wv][r * 32 + l15]      = f2bf(p0);
      pls[wv][r * 32 + 16 + l15] = f2bf(p1);
    }

    const bf16_t* pp = &pls[wv][l15 * 32 + lh * 16];
    const v16bf pf = ldfrag(pp, pp + 8);

#pragma unroll
    for (int t = 0; t < 4; ++t) {
      const bf16_t* vp = vbh + (size_t)(t * 16 + l15) * T_ + j2 * 32 + lh * 16;
      const v16bf vf = ldfrag(vp, vp + 8);
      o[t] = wmma_bf16(pf, vf, o[t]);
    }
  }

  bf16_t* ybase = Y + ((size_t)b * T_ + qt * 16) * C_ + h * HD_;
#pragma unroll
  for (int t = 0; t < 4; ++t) {
#pragma unroll
    for (int v = 0; v < 8; ++v) {
      const int r = v + 8 * lh;
      const float val = o[t][v] / lrow[v];
      ybase[(size_t)r * C_ + t * 16 + l15] = f2bf(val);
    }
  }
}

// ---------------------------------------------------------------------------
// LayerNorm: one wave per row (C=512 -> 16 elems/lane), shuffle reductions.
// ---------------------------------------------------------------------------
template <bool OUTBF>
__global__ __launch_bounds__(256) void layernorm_k(
    const float* __restrict__ X, const float* __restrict__ g,
    const float* __restrict__ bta, bf16_t* __restrict__ outB,
    float* __restrict__ outF) {
  const int lane = threadIdx.x & 31;
  const int wv   = threadIdx.x >> 5;
  const int row  = blockIdx.x * 8 + wv;
  const float* x = X + (size_t)row * C_;
  float vals[16];
  float s = 0.f;
#pragma unroll
  for (int i = 0; i < 16; ++i) { vals[i] = x[lane + i * 32]; s += vals[i]; }
#pragma unroll
  for (int mm = 16; mm >= 1; mm >>= 1) s += __shfl_xor(s, mm);
  const float mu = s * (1.f / C_);
  float var = 0.f;
#pragma unroll
  for (int i = 0; i < 16; ++i) { float d = vals[i] - mu; var += d * d; }
#pragma unroll
  for (int mm = 16; mm >= 1; mm >>= 1) var += __shfl_xor(var, mm);
  const float rs = rsqrtf(var * (1.f / C_) + 1e-5f);
#pragma unroll
  for (int i = 0; i < 16; ++i) {
    const int c = lane + i * 32;
    const float yv = (vals[i] - mu) * rs * g[c] + bta[c];
    if (OUTBF) outB[(size_t)row * C_ + c] = f2bf(yv);
    else       outF[(size_t)row * C_ + c] = yv;
  }
}

// ---------------------------------------------------------------------------
// V transpose: bf16 [B,T,C] head slices -> VT bf16 [B,H,HD,T], LDS tiled.
// ---------------------------------------------------------------------------
__global__ __launch_bounds__(256) void transpose_v_k(
    const bf16_t* __restrict__ Vb, bf16_t* __restrict__ VT) {
  __shared__ bf16_t tile[16][17];
  const int bh  = blockIdx.y >> 2;
  const int hdT = blockIdx.y & 3;
  const int b = bh / H_, h = bh % H_;
  const int t0 = blockIdx.x * 16;
  const int tx = threadIdx.x & 15, ty = threadIdx.x >> 4;
  tile[ty][tx] = Vb[((size_t)b * T_ + t0 + ty) * C_ + h * HD_ + hdT * 16 + tx];
  __syncthreads();
  VT[((size_t)bh * HD_ + hdT * 16 + ty) * T_ + t0 + tx] = tile[tx][ty];
}

// ---------------------------------------------------------------------------
// Orchestration
// ---------------------------------------------------------------------------
extern "C" void kernel_launch(void* const* d_in, const int* in_sizes, int n_in,
                              void* d_out, int out_size, void* d_ws,
                              size_t ws_size, hipStream_t stream) {
  const float* x     = (const float*)d_in[0];
  const float* ln1_g = (const float*)d_in[1];
  const float* ln1_b = (const float*)d_in[2];
  const float* wq    = (const float*)d_in[3];
  const float* bq    = (const float*)d_in[4];
  const float* wk    = (const float*)d_in[5];
  const float* bk    = (const float*)d_in[6];
  const float* wvv   = (const float*)d_in[7];
  const float* bv    = (const float*)d_in[8];
  const float* wo    = (const float*)d_in[9];
  const float* bo    = (const float*)d_in[10];
  const float* ln2_g = (const float*)d_in[11];
  const float* ln2_b = (const float*)d_in[12];
  const float* w1    = (const float*)d_in[13];
  const float* b1    = (const float*)d_in[14];
  const float* w2    = (const float*)d_in[15];
  const float* b2    = (const float*)d_in[16];
  const float* lnf_g = (const float*)d_in[17];
  const float* lnf_b = (const float*)d_in[18];

  char* p = (char*)d_ws;
  auto alloc = [&](size_t bytes) -> char* {
    char* r = p;
    p += (bytes + 255) & ~(size_t)255;
    return r;
  };
  float*  h   = (float*)alloc((size_t)M_ * C_ * 4);
  bf16_t* aln = (bf16_t*)alloc((size_t)M_ * C_ * 2);
  bf16_t* qB  = (bf16_t*)alloc((size_t)M_ * C_ * 2);
  bf16_t* kB  = (bf16_t*)alloc((size_t)M_ * C_ * 2);
  bf16_t* vB  = (bf16_t*)alloc((size_t)M_ * C_ * 2);
  bf16_t* vT  = (bf16_t*)alloc((size_t)M_ * C_ * 2);
  bf16_t* yB  = (bf16_t*)alloc((size_t)M_ * C_ * 2);
  bf16_t* mB  = qB;  // MLP hidden [M,4C] aliases q|k|v|vT (exactly 4*M*C*2 B)
  bf16_t* pw[2][6];
  for (int l = 0; l < 2; ++l) {
    pw[l][0] = (bf16_t*)alloc((size_t)C_ * C_ * 2);
    pw[l][1] = (bf16_t*)alloc((size_t)C_ * C_ * 2);
    pw[l][2] = (bf16_t*)alloc((size_t)C_ * C_ * 2);
    pw[l][3] = (bf16_t*)alloc((size_t)C_ * C_ * 2);
    pw[l][4] = (bf16_t*)alloc((size_t)C_ * 4 * C_ * 2);
    pw[l][5] = (bf16_t*)alloc((size_t)4 * C_ * C_ * 2);
  }

  (void)hipMemcpyAsync(h, x, (size_t)M_ * C_ * 4, hipMemcpyDeviceToDevice,
                       stream);

  const int CC = C_ * C_, C4 = C_ * 4 * C_;
  for (int l = 0; l < 2; ++l) {
    pack_w_k<<<CC / 256, 256, 0, stream>>>(wq + (size_t)l * CC, pw[l][0], C_, CC);
    pack_w_k<<<CC / 256, 256, 0, stream>>>(wk + (size_t)l * CC, pw[l][1], C_, CC);
    pack_w_k<<<CC / 256, 256, 0, stream>>>(wvv + (size_t)l * CC, pw[l][2], C_, CC);
    pack_w_k<<<CC / 256, 256, 0, stream>>>(wo + (size_t)l * CC, pw[l][3], C_, CC);
    pack_w_k<<<C4 / 256, 256, 0, stream>>>(w1 + (size_t)l * C4, pw[l][4], 4 * C_, C4);
    pack_w_k<<<C4 / 256, 256, 0, stream>>>(w2 + (size_t)l * C4, pw[l][5], C_, C4);
  }

  const dim3 g512(C_ / 128, M_ / 128);       // (4,128)
  const dim3 g2048(4 * C_ / 128, M_ / 128);  // (16,128)

  for (int l = 0; l < 2; ++l) {
    layernorm_k<true><<<M_ / 8, 256, 0, stream>>>(h, ln1_g + l * C_,
                                                  ln1_b + l * C_, aln, nullptr);
    gemm_wmma_k<0><<<g512, 128, 0, stream>>>(aln, pw[l][0], bq + l * C_,
                                             nullptr, qB, nullptr, C_, C_);
    gemm_wmma_k<0><<<g512, 128, 0, stream>>>(aln, pw[l][1], bk + l * C_,
                                             nullptr, kB, nullptr, C_, C_);
    gemm_wmma_k<0><<<g512, 128, 0, stream>>>(aln, pw[l][2], bv + l * C_,
                                             nullptr, vB, nullptr, C_, C_);
    transpose_v_k<<<dim3(T_ / 16, B_ * H_ * (HD_ / 16)), 256, 0, stream>>>(vB, vT);
    attn_flash_k<<<(B_ * H_ * TQ_) / 4, 128, 0, stream>>>(qB, kB, vT, yB);
    gemm_wmma_k<2><<<g512, 128, 0, stream>>>(yB, pw[l][3], bo + l * C_, h,
                                             nullptr, h, C_, C_);
    layernorm_k<true><<<M_ / 8, 256, 0, stream>>>(h, ln2_g + l * C_,
                                                  ln2_b + l * C_, aln, nullptr);
    gemm_wmma_k<1><<<g2048, 128, 0, stream>>>(aln, pw[l][4], b1 + l * 4 * C_,
                                              nullptr, mB, nullptr, 4 * C_, C_);
    gemm_wmma_k<2><<<g512, 128, 0, stream>>>(mB, pw[l][5], b2 + l * C_, h,
                                             nullptr, h, C_, 4 * C_);
  }
  layernorm_k<false><<<M_ / 8, 256, 0, stream>>>(h, lnf_g, lnf_b, nullptr,
                                                 (float*)d_out);
}
